// SDEABlock_89824946029194
// MI455X (gfx1250) — matrix operand order repaired and checked by reading, not compiled, not run on hardware
//
#include <hip/hip_runtime.h>
#include <hip/hip_bf16.h>

// ---------------------------------------------------------------------------
// SDEA block for MI455X (gfx1250): conv3x3 via V_WMMA_F32_16X16X32_BF16 with
// b128 LDS fragment loads, fused per-channel BN stats (ds_add_f32 +
// global_atomic_add_f32), bf16 global activations, async-to-LDS weight
// staging, disparity search, residual fuse.
// ---------------------------------------------------------------------------

typedef __attribute__((ext_vector_type(16))) __bf16 v16bf;
typedef __attribute__((ext_vector_type(8)))  __bf16 v8bf;
typedef __attribute__((ext_vector_type(4)))  __bf16 v4bf;
typedef __attribute__((ext_vector_type(8)))  float  v8f;
typedef __attribute__((ext_vector_type(4)))  int    v4i;

#define Bn     4
#define Cc     32
#define Hh     256
#define Ww     512
#define HW     (Hh * Ww)                // 131072
#define NELEM  (Bn * Cc * HW)           // 16777216
#define NPIX   (Bn * HW)                // 524288
#define TILE_X 128                      // pixels per block (8 waves x 16)
#define LDSW   132                      // 130 halo cols (padded tile x extent)
#define NTAP   9
#define CPAD   40                       // padded channel stride (bf16 elems, 80B)

#if __has_builtin(__builtin_amdgcn_global_load_async_to_lds_b128)
#define HAVE_ASYNC_LDS 1
// Builtin expects (v4i addrspace(1)*, v4i addrspace(3)*, imm offset, imm cpol).
#define AS1V(p) ((__attribute__((address_space(1))) v4i*)(p))
#define AS3V(p) ((__attribute__((address_space(3))) v4i*)(p))
#else
#define HAVE_ASYNC_LDS 0
#endif

// ---------------------------------------------------------------------------
// f32 NCHW -> bf16 NCHW (one pass; conv inputs read half the bytes).
// ---------------------------------------------------------------------------
__global__ __launch_bounds__(256) void f32_to_bf16(
    const float* __restrict__ x, __bf16* __restrict__ y)
{
    const int idx = blockIdx.x * 256 + threadIdx.x;   // float4 index
    float4 v = ((const float4*)x)[idx];
    v4bf o;
    o[0] = (__bf16)v.x; o[1] = (__bf16)v.y; o[2] = (__bf16)v.z; o[3] = (__bf16)v.w;
    ((v4bf*)y)[idx] = o;
}

// ---------------------------------------------------------------------------
// Weight prep: f32 OIHW [k][c][dy][dx] -> bf16 [tap][k][c] (c contiguous).
// ---------------------------------------------------------------------------
__global__ __launch_bounds__(256) void prep_w(
    const float* __restrict__ w, __bf16* __restrict__ wt)
{
    const int i = blockIdx.x * 256 + threadIdx.x;
    if (i >= NTAP * Cc * Cc) return;
    const int tap = i / (Cc * Cc);
    const int r   = i - tap * (Cc * Cc);
    const int k   = r >> 5;
    const int c   = r & 31;
    wt[i] = (__bf16)w[(k * Cc + c) * NTAP + tap];
}

// ---------------------------------------------------------------------------
// conv3x3, C=32 -> K=32, pad 1, implicit GEMM on WMMA bf16 tensor path,
// with fused per-channel BN-stat accumulation.
// LDS layouts are K-innermost so every WMMA fragment is 2x ds_load_b128:
//   sIn[row][x][c]  (c stride padded to 40 -> 80B lane stride, 16B aligned)
//   sW [tap][kout][c]
// Staging reads global in aligned 16B (v8bf) chunks; per-channel partial
// sums go LDS ds_add_f32 -> one global_atomic_add_f32 per channel per block.
// ---------------------------------------------------------------------------
__global__ __launch_bounds__(256) void conv3x3_bf16_wmma(
    const __bf16* __restrict__ in, const __bf16* __restrict__ wt,
    float* __restrict__ out, float* __restrict__ stats)
{
    __shared__ __align__(16) __bf16 sIn[3 * LDSW * CPAD];  // 31680 B
    __shared__ __align__(16) __bf16 sW[NTAP * Cc * CPAD];  // 23040 B
    __shared__ float sStat[64];                            // 32 sum + 32 sumsq

    const int bid = blockIdx.x;
    const int xt  = bid & 3;                 // W / TILE_X = 4
    const int y   = (bid >> 2) & (Hh - 1);
    const int b   = bid >> 10;
    const int x0  = xt * TILE_X;
    const int tid = threadIdx.x;

    __builtin_prefetch(wt, 0, 0);
    if (tid < 64) sStat[tid] = 0.f;

    // --- Stage weights [tap][k][c]: contiguous 64B rows -> padded LDS rows.
#if HAVE_ASYNC_LDS
    // Per-lane async DMA: each lane moves one aligned 16B chunk straight into
    // LDS, no VGPR round trip. 9*32 rows x 4 chunks = 1152 chunks.
    for (int ch = tid; ch < NTAP * Cc * 4; ch += 256) {
        const int r = ch >> 2;
        const int q = ch & 3;
        __builtin_amdgcn_global_load_async_to_lds_b128(
            AS1V(wt + r * Cc + q * 8), AS3V(sW + r * CPAD + q * 8), 0, 0);
    }
#else
    for (int i = tid; i < NTAP * Cc * Cc; i += 256) {
        const int r = i >> 5;
        const int c = i & 31;
        sW[r * CPAD + c] = wt[i];
    }
#endif

    // --- Stage input halo tile (3 rows x 130 cols x 32 ch), zero-padded,
    //     transposed to channel-innermost. Global reads: aligned v8bf chunks
    //     on global-x granularity (chunks at gx0 = x0-8+8j, j=0..17 cover the
    //     halo range [x0-1, x0+129)); LDS writes scatter the transpose.
    for (int t = tid; t < Cc * 3 * 18; t += 256) {
        const int c  = t / (3 * 18);
        const int r  = t - c * (3 * 18);
        const int ry = r / 18;
        const int j  = r - ry * 18;
        const int gy  = y + ry - 1;
        const int gx0 = x0 - 8 + j * 8;
        v8bf v = {};
        if ((unsigned)gy < (unsigned)Hh && (unsigned)gx0 < (unsigned)Ww)
            v = *(const v8bf*)(in + ((b * Cc + c) * Hh + gy) * Ww + gx0);
#pragma unroll
        for (int e = 0; e < 8; ++e) {
            const int rx = gx0 + e - x0 + 1;
            if ((unsigned)rx < 130u)
                sIn[(ry * LDSW + rx) * CPAD + c] = v[e];
        }
    }

#if HAVE_ASYNC_LDS
#if __has_builtin(__builtin_amdgcn_s_wait_asynccnt)
    __builtin_amdgcn_s_wait_asynccnt(0);
#else
    asm volatile("s_wait_asynccnt 0" ::: "memory");
#endif
#endif
    __syncthreads();

    const int lane  = tid & 31;
    const int lmod  = lane & 15;
    const int lhalf = lane >> 4;
    const int px    = (tid >> 5) * 16;       // wave pixel offset in tile

    v8f acc0 = {};
    v8f acc1 = {};

#pragma unroll
    for (int tap = 0; tap < NTAP; ++tap) {
        const int dy = tap / 3;
        const int dx = tap - dy * 3;

        // A fragment (16x32 bf16): lane holds M=lmod; K halfwords ascend as
        // c = lhalf*16 + {0..15}  -> 2x b128.
        const __bf16* ap = &sIn[((dy * LDSW) + px + lmod + dx) * CPAD + (lhalf << 4)];
        const v8bf alo = *(const v8bf*)ap;
        const v8bf ahi = *(const v8bf*)(ap + 8);
        const v16bf a = __builtin_shufflevector(alo, ahi,
            0, 1, 2, 3, 4, 5, 6, 7, 8, 9, 10, 11, 12, 13, 14, 15);

        // B fragments (32x16 bf16): lane holds N=lmod (+16 for second tile),
        // K = lhalf*16 + {0..15} ascending -> 2x b128 each.
        const __bf16* bp0 = &sW[(tap * Cc + lmod) * CPAD + (lhalf << 4)];
        const v8bf b0lo = *(const v8bf*)bp0;
        const v8bf b0hi = *(const v8bf*)(bp0 + 8);
        const v16bf b0 = __builtin_shufflevector(b0lo, b0hi,
            0, 1, 2, 3, 4, 5, 6, 7, 8, 9, 10, 11, 12, 13, 14, 15);

        const __bf16* bp1 = bp0 + 16 * CPAD;
        const v8bf b1lo = *(const v8bf*)bp1;
        const v8bf b1hi = *(const v8bf*)(bp1 + 8);
        const v16bf b1 = __builtin_shufflevector(b1lo, b1hi,
            0, 1, 2, 3, 4, 5, 6, 7, 8, 9, 10, 11, 12, 13, 14, 15);

        acc0 = __builtin_amdgcn_wmma_f32_16x16x32_bf16(
            false, a, false, b0, (short)0, acc0, false, false);
        acc1 = __builtin_amdgcn_wmma_f32_16x16x32_bf16(
            false, a, false, b1, (short)0, acc1, false, false);
    }

    // D layout: VGPR r -> M = r + 8*lhalf, N = lmod (acc0: kout=lmod,
    // acc1: kout=16+lmod). Store + fused per-channel stat reduction.
    float s0 = 0.f, q0 = 0.f, s1 = 0.f, q1 = 0.f;
#pragma unroll
    for (int r = 0; r < 8; ++r) {
        const int gx = x0 + px + r + (lhalf << 3);
        out[((b * Cc + lmod) * Hh + y) * Ww + gx]      = acc0[r];
        out[((b * Cc + 16 + lmod) * Hh + y) * Ww + gx] = acc1[r];
        s0 += acc0[r]; q0 += acc0[r] * acc0[r];
        s1 += acc1[r]; q1 += acc1[r] * acc1[r];
    }
    atomicAdd(&sStat[lmod],      s0);
    atomicAdd(&sStat[32 + lmod], q0);
    atomicAdd(&sStat[16 + lmod], s1);
    atomicAdd(&sStat[48 + lmod], q1);
    __syncthreads();
    if (tid < 64) atomicAdd(&stats[tid], sStat[tid]);
}

__global__ void zero_stats(float* __restrict__ stats)
{
    stats[threadIdx.x] = 0.f;   // 64 threads: 32 sums + 32 sumsq
}

// ---------------------------------------------------------------------------
// Fused BN + ReLU -> bf16 output (feeds the next conv's tensor path).
// ---------------------------------------------------------------------------
__global__ __launch_bounds__(256) void bn_relu_bf16(
    const float* __restrict__ x, const float* __restrict__ stats,
    const float* __restrict__ gamma, const float* __restrict__ beta,
    __bf16* __restrict__ y)
{
    __shared__ float sc[Cc];
    __shared__ float sh[Cc];
    const int tid = threadIdx.x;
    if (tid < Cc) {
        const float invN = 1.f / (float)(Bn * HW);
        const float mean = stats[tid] * invN;
        const float var  = stats[Cc + tid] * invN - mean * mean;
        const float s    = rsqrtf(var + 1e-5f) * gamma[tid];
        sc[tid] = s;
        sh[tid] = beta[tid] - mean * s;
    }
    __syncthreads();

    const int idx  = blockIdx.x * 256 + tid;  // float4 index
    const int c    = ((idx * 4) / HW) & (Cc - 1);
    float4 v = ((const float4*)x)[idx];
    const float s = sc[c], t = sh[c];
    v4bf o;
    o[0] = (__bf16)fmaxf(v.x * s + t, 0.f);
    o[1] = (__bf16)fmaxf(v.y * s + t, 0.f);
    o[2] = (__bf16)fmaxf(v.z * s + t, 0.f);
    o[3] = (__bf16)fmaxf(v.w * s + t, 0.f);
    ((v4bf*)y)[idx] = o;
}

// ---------------------------------------------------------------------------
// Fused BN + ReLU -> f32 output (g1, consumed by g2/fuse).
// ---------------------------------------------------------------------------
__global__ __launch_bounds__(256) void bn_relu_f32(
    const float* __restrict__ x, const float* __restrict__ stats,
    const float* __restrict__ gamma, const float* __restrict__ beta,
    float* __restrict__ y)
{
    __shared__ float sc[Cc];
    __shared__ float sh[Cc];
    const int tid = threadIdx.x;
    if (tid < Cc) {
        const float invN = 1.f / (float)(Bn * HW);
        const float mean = stats[tid] * invN;
        const float var  = stats[Cc + tid] * invN - mean * mean;
        const float s    = rsqrtf(var + 1e-5f) * gamma[tid];
        sc[tid] = s;
        sh[tid] = beta[tid] - mean * s;
    }
    __syncthreads();

    const int idx  = blockIdx.x * 256 + tid;
    const int c    = ((idx * 4) / HW) & (Cc - 1);
    float4 v = ((const float4*)x)[idx];
    const float s = sc[c], t = sh[c];
    v.x = fmaxf(v.x * s + t, 0.f);
    v.y = fmaxf(v.y * s + t, 0.f);
    v.z = fmaxf(v.z * s + t, 0.f);
    v.w = fmaxf(v.w * s + t, 0.f);
    ((float4*)y)[idx] = v;
}

// ---------------------------------------------------------------------------
// 1x1 conv 32 -> 1 (+bias): per-pixel dot over channels.
// ---------------------------------------------------------------------------
__global__ __launch_bounds__(256) void g2_conv(
    const float* __restrict__ g1, const float* __restrict__ gw,
    const float* __restrict__ gb, float* __restrict__ outm)
{
    const int p = blockIdx.x * 256 + threadIdx.x;
    if (p >= NPIX) return;
    const int bb = p / HW;
    const int r  = p - bb * HW;
    float s = gb[0];
#pragma unroll
    for (int c = 0; c < Cc; ++c)
        s += g1[(bb * Cc + c) * HW + r] * gw[c];
    outm[p] = s;
}

// ---------------------------------------------------------------------------
// Disparity weight: min over d in [-md, md) of |R[x+d]-L[x]|; 1/(1+e^min).
// ---------------------------------------------------------------------------
__global__ __launch_bounds__(256) void wvol(
    const float* __restrict__ Lm, const float* __restrict__ Rm,
    const int* __restrict__ mdp, float* __restrict__ wout)
{
    const int p = blockIdx.x * 256 + threadIdx.x;
    if (p >= NPIX) return;
    const int md = mdp[0];
    const int x  = p & (Ww - 1);
    const int rowbase = p - x;
    const float L = Lm[p];
    float best = 3.402823466e38f;
    for (int d = -md; d < md; ++d) {
        const int ix = x + d;
        if ((unsigned)ix < (unsigned)Ww)
            best = fminf(best, fabsf(Rm[rowbase + ix] - L));
    }
    wout[p] = 1.f / (1.f + __expf(best));
}

// ---------------------------------------------------------------------------
// out = g1 * weight(broadcast over C) + input, float4-vectorized.
// ---------------------------------------------------------------------------
__global__ __launch_bounds__(256) void fuse_out(
    const float* __restrict__ g1, const float* __restrict__ wmap,
    const float* __restrict__ inp, float* __restrict__ out)
{
    const int idx  = blockIdx.x * 256 + threadIdx.x;  // float4 index
    const int base = idx * 4;
    const int bb   = base / (Cc * HW);
    const int pi   = base & (HW - 1);
    const float4 g = ((const float4*)g1)[idx];
    const float4 w = ((const float4*)wmap)[(bb * HW + pi) >> 2];
    const float4 i = ((const float4*)inp)[idx];
    float4 o;
    o.x = g.x * w.x + i.x;
    o.y = g.y * w.y + i.y;
    o.z = g.z * w.z + i.z;
    o.w = g.w * w.w + i.w;
    ((float4*)out)[idx] = o;
}

// ---------------------------------------------------------------------------
extern "C" void kernel_launch(void* const* d_in, const int* in_sizes, int n_in,
                              void* d_out, int out_size, void* d_ws, size_t ws_size,
                              hipStream_t stream)
{
    const float* left  = (const float*)d_in[0];
    const float* right = (const float*)d_in[1];
    const float* w1    = (const float*)d_in[2];
    const float* bn1g  = (const float*)d_in[3];
    const float* bn1b  = (const float*)d_in[4];
    const float* w2    = (const float*)d_in[5];
    const float* bn2g  = (const float*)d_in[6];
    const float* bn2b  = (const float*)d_in[7];
    const float* g2w   = (const float*)d_in[8];
    const float* g2b   = (const float*)d_in[9];
    const int*   mdp   = (const int*)d_in[10];

    float* ws   = (float*)d_ws;
    float* A    = ws;                               // conv output scratch (f32)
    float* G1L  = ws + (size_t)NELEM;
    float* G1R  = ws + 2 * (size_t)NELEM;
    __bf16* X16 = (__bf16*)(ws + 3 * (size_t)NELEM); // bf16 activations (NELEM)
    float* tail = ws + 3 * (size_t)NELEM + (size_t)NELEM / 2;
    float* G2L   = tail;
    float* G2R   = G2L + NPIX;
    float* WL    = G2R + NPIX;
    float* WR    = WL + NPIX;
    float* STATS = WR + NPIX;                       // 64 floats
    __bf16* WT1  = (__bf16*)(STATS + 64);           // 9216 bf16
    __bf16* WT2  = WT1 + NTAP * Cc * Cc;

    const dim3 convGrid(Bn * Hh * (Ww / TILE_X));   // 4096 blocks
    const dim3 ewGrid(NELEM / 1024);                // float4 elementwise
    const dim3 pixGrid(NPIX / 256);

    prep_w<<<36, 256, 0, stream>>>(w1, WT1);
    prep_w<<<36, 256, 0, stream>>>(w2, WT2);

    auto side = [&](const float* inp, float* g1) {
        f32_to_bf16<<<ewGrid, 256, 0, stream>>>(inp, X16);
        zero_stats<<<1, 64, 0, stream>>>(STATS);
        conv3x3_bf16_wmma<<<convGrid, 256, 0, stream>>>(X16, WT1, A, STATS);
        bn_relu_bf16<<<ewGrid, 256, 0, stream>>>(A, STATS, bn1g, bn1b, X16);
        zero_stats<<<1, 64, 0, stream>>>(STATS);
        conv3x3_bf16_wmma<<<convGrid, 256, 0, stream>>>(X16, WT2, A, STATS);
        bn_relu_f32<<<ewGrid, 256, 0, stream>>>(A, STATS, bn2g, bn2b, g1);
    };
    side(left,  G1L);
    side(right, G1R);

    g2_conv<<<pixGrid, 256, 0, stream>>>(G1L, g2w, g2b, G2L);
    g2_conv<<<pixGrid, 256, 0, stream>>>(G1R, g2w, g2b, G2R);
    wvol<<<pixGrid, 256, 0, stream>>>(G2L, G2R, mdp, WL);
    wvol<<<pixGrid, 256, 0, stream>>>(G2R, G2L, mdp, WR);

    float* out = (float*)d_out;
    fuse_out<<<ewGrid, 256, 0, stream>>>(G1L, WL, left,  out);
    fuse_out<<<ewGrid, 256, 0, stream>>>(G1R, WR, right, out + (size_t)NELEM);
}
